// ModellingLayer_82867099009930
// MI455X (gfx1250) — matrix operand order repaired and checked by reading, not compiled, not run on hardware
//
#include <hip/hip_runtime.h>
#include <hip/hip_bf16.h>

// CDNA5 wave32 WMMA types
typedef __attribute__((ext_vector_type(16))) __bf16 v16bf;
typedef __attribute__((ext_vector_type(8)))  __bf16 v8bf;
typedef __attribute__((ext_vector_type(8)))  float  v8f;

__device__ __forceinline__ float sigf(float x) { return 1.0f / (1.0f + __expf(-x)); }

// ---------------------------------------------------------------------------
// CDNA5 async global->LDS (ASYNCcnt) with sync fallback.
// Builtin signature (from hipcc diagnostic): param 1 is
//   'int __attribute__((vector_size(16))) __device__ *'  (AS1 int4*)
// so cast to AS-qualified int4 pointee types.
// ---------------------------------------------------------------------------
#if defined(__has_builtin)
# if __has_builtin(__builtin_amdgcn_global_load_async_to_lds_b128)
#  define HAVE_ASYNC_LDS 1
# else
#  define HAVE_ASYNC_LDS 0
# endif
# if __has_builtin(__builtin_amdgcn_s_wait_asynccnt)
#  define HAVE_WAIT_ASYNC 1
# else
#  define HAVE_WAIT_ASYNC 0
# endif
#else
# define HAVE_ASYNC_LDS 0
# define HAVE_WAIT_ASYNC 0
#endif

typedef int v4i_vs __attribute__((vector_size(16)));
typedef __attribute__((address_space(1))) v4i_vs as1_v4i;
typedef __attribute__((address_space(3))) v4i_vs as3_v4i;

__device__ __forceinline__ void async_cp16(const __bf16* g, __bf16* l) {
#if HAVE_ASYNC_LDS
  __builtin_amdgcn_global_load_async_to_lds_b128(
      (as1_v4i*)g, (as3_v4i*)l, 0, 0);
#else
  *(v8bf*)l = *(const v8bf*)g;
#endif
}

__device__ __forceinline__ void async_wait0() {
#if HAVE_ASYNC_LDS
# if HAVE_WAIT_ASYNC
  __builtin_amdgcn_s_wait_asynccnt(0);
# else
  asm volatile("s_wait_asynccnt 0x0" ::: "memory");
# endif
#endif
}

// ---------------------------------------------------------------------------
// Problem constants
// ---------------------------------------------------------------------------
#define TT 8192
#define DD 2048
#define HH 256      // hidden
#define G4 1024     // 4*H
#define HOUT 2560   // D + 2H (output row width)

// ---------------------------------------------------------------------------
// f32 -> bf16 convert (grid-stride)
// ---------------------------------------------------------------------------
__global__ void cvt_f32_to_bf16(const float* __restrict__ src,
                                __bf16* __restrict__ dst, int n) {
  int i = blockIdx.x * blockDim.x + threadIdx.x;
  int stride = gridDim.x * blockDim.x;
  for (; i < n; i += stride) dst[i] = (__bf16)src[i];
}

// W (K x N row-major) -> Wt (N x K row-major) bf16.  Output-linear indexing.
__global__ void cvt_W_transpose(const float* __restrict__ W,
                                __bf16* __restrict__ Wt, int K, int N) {
  int i = blockIdx.x * blockDim.x + threadIdx.x;
  int stride = gridDim.x * blockDim.x;
  int total = K * N;
  for (; i < total; i += stride) {
    int n = i / K, k = i - n * K;
    Wt[i] = (__bf16)W[(size_t)k * N + n];
  }
}

// U (256 x 1024, row-major, gate order i,f,g,o each 256 cols) ->
// per-half, gate-major, N-major bf16: dst[(hb*512 + c)*256 + k]
//   c = g*128 + jj  maps to U column g*256 + hb*128 + jj
__global__ void cvt_U_perm(const float* __restrict__ U, __bf16* __restrict__ dst) {
  int i = blockIdx.x * blockDim.x + threadIdx.x;
  int stride = gridDim.x * blockDim.x;
  const int total = 2 * 512 * 256;
  for (; i < total; i += stride) {
    int k  = i & 255;
    int c  = (i >> 8) & 511;
    int hb = i >> 17;
    int g  = c >> 7;
    int jj = c & 127;
    dst[i] = (__bf16)U[(size_t)k * G4 + g * HH + hb * 128 + jj];
  }
}

// ---------------------------------------------------------------------------
// bf16 WMMA GEMM:  C(M,N) f32 = A(M,K)bf16 @ Bt(N,K)bf16^T + bias[N]
// Both operands row-major along K -> both tiles staged with async b128 DMA
// into double-buffered LDS; async copy of tile k+1 overlaps WMMA on tile k.
// Block tile 128x128, BK=32. 256 threads = 8 waves (4M x 2N); each wave owns
// a 32x64 tile = 2x4 WMMA 16x16x32 tiles.
// ---------------------------------------------------------------------------
#define GBM 128
#define GBN 128
#define GBK 32
#define LDT 48   // LDS row pitch in bf16 elems (16B aligned, breaks bank period)

__global__ __launch_bounds__(256)
void gemm_bf16_wmma(const __bf16* __restrict__ A, int lda,
                    const __bf16* __restrict__ Bt, int ldbt,
                    const float* __restrict__ bias,
                    float* __restrict__ C, int ldc, int K) {
  __shared__ __bf16 As[2][GBM * LDT];   // [row m][k]
  __shared__ __bf16 Bs[2][GBN * LDT];   // [col n][k]

  const int tid  = threadIdx.x;
  const int lane = tid & 31;
  const int w    = tid >> 5;      // 0..7
  const int mB   = (w >> 1) * 32; // wave M origin in tile
  const int nB   = (w & 1) * 64;  // wave N origin in tile
  const size_t mBlock = (size_t)blockIdx.y * GBM;
  const size_t nBlock = (size_t)blockIdx.x * GBN;
  const int kg = lane >> 4;

  v8f acc[2][4];
  for (int i = 0; i < 2; ++i)
    for (int j = 0; j < 4; ++j)
      for (int e = 0; e < 8; ++e) acc[i][j][e] = 0.0f;

  // issue async DMA of one 128x32 A tile + 128x32 B tile into buffer `buf`
  auto issue = [&](int buf, int k0) {
#pragma unroll
    for (int it = 0; it < 2; ++it) {
      int idx = tid * 2 + it;
      int row = idx >> 2, ch = idx & 3;           // 512 chunks of 8 bf16
      async_cp16(&A[(mBlock + row) * (size_t)lda + k0 + ch * 8],
                 &As[buf][row * LDT + ch * 8]);
      async_cp16(&Bt[(nBlock + row) * (size_t)ldbt + k0 + ch * 8],
                 &Bs[buf][row * LDT + ch * 8]);
    }
  };

  issue(0, 0);
  int buf = 0;
  for (int k0 = 0; k0 < K; k0 += GBK) {
    async_wait0();        // own async issues (current buf) complete
    __syncthreads();      // -> whole tile visible; prev compute done
    if (k0 + GBK < K) {
      issue(buf ^ 1, k0 + GBK);                  // overlap DMA with WMMA
      __builtin_prefetch(&A[(mBlock + tid / 2) * (size_t)lda + k0 + 2 * GBK], 0, 1);
    }

    union { v16bf v; v8bf h[2]; } a[2], b[4];
#pragma unroll
    for (int i = 0; i < 2; ++i) {
      int row = mB + i * 16 + (lane & 15);
      a[i].h[0] = *(const v8bf*)&As[buf][row * LDT + kg * 8];      // K 0-7 / 8-15
      a[i].h[1] = *(const v8bf*)&As[buf][row * LDT + 16 + kg * 8]; // K 16-23 / 24-31
    }
#pragma unroll
    for (int j = 0; j < 4; ++j) {
      int col = nB + j * 16 + (lane & 15);
      b[j].h[0] = *(const v8bf*)&Bs[buf][col * LDT + kg * 8];
      b[j].h[1] = *(const v8bf*)&Bs[buf][col * LDT + 16 + kg * 8];
    }
#pragma unroll
    for (int i = 0; i < 2; ++i)
#pragma unroll
      for (int j = 0; j < 4; ++j)
        acc[i][j] = __builtin_amdgcn_wmma_f32_16x16x32_bf16(
            false, a[i].v, false, b[j].v, (short)0, acc[i][j], false, false);
    buf ^= 1;
  }

  // epilogue: VGPR r -> M = r + 8*(lane>>4), N = lane&15
#pragma unroll
  for (int i = 0; i < 2; ++i)
#pragma unroll
    for (int j = 0; j < 4; ++j) {
      size_t col = nBlock + nB + j * 16 + (lane & 15);
      float bv = bias[col];
#pragma unroll
      for (int r = 0; r < 8; ++r) {
        size_t row = mBlock + mB + i * 16 + r + 8 * (lane >> 4);
        C[row * (size_t)ldc + col] = acc[i][j][r] + bv;
      }
    }
}

// ---------------------------------------------------------------------------
// Persistent LSTM recurrence. Grid = 4 blocks: blockIdx.x = dir*2 + half.
// Each block holds U[:, its 512 gate-cols] as bf16 (256x512 -> 256KB LDS,
// N-major with pad, staged via async DMA), owns 128 hidden units, computes
// h@U with WMMA by replicating h across the 16 A rows and extracting row M=0.
// h halves are exchanged through L2 with a flag handshake per step.
// ---------------------------------------------------------------------------
#define REC_UPITCH 264  // 256 + 8 pad (528B row stride -> no bank-period hit)

__global__ __launch_bounds__(256)
void lstm_rec(const float* __restrict__ zxF, const float* __restrict__ zxB,
              const __bf16* __restrict__ UpermAll,   // 4 slices of 512*256
              float* __restrict__ outPlane, int outStride, int colOff0,
              __bf16* __restrict__ outBf, int bfStride,   // may be null
              __bf16* __restrict__ hx, unsigned* __restrict__ flags, int T) {
  extern __shared__ char smem[];
  __bf16* Us = (__bf16*)smem;                              // [512][264]
  __bf16* hs = (__bf16*)(smem + 512 * REC_UPITCH * 2);     // [256]
  float*  zr = (float*) (smem + 512 * REC_UPITCH * 2 + 512); // [512]

  const int tid  = threadIdx.x;
  const int lane = tid & 31;
  const int w    = tid >> 5;
  const int dir  = blockIdx.x >> 1;
  const int half = blockIdx.x & 1;
  const int me = dir * 2 + half, other = dir * 2 + (1 - half);
  const float* zx = dir ? zxB : zxF;
  const __bf16* Ug = UpermAll + (size_t)me * (512 * 256);
  const int kg = lane >> 4;

  // stage U slice into LDS (256KB) via async DMA
  for (int idx = tid; idx < 512 * 32; idx += 256) {
    int row = idx >> 5, ch = idx & 31;
    async_cp16(&Ug[row * 256 + ch * 8], &Us[row * REC_UPITCH + ch * 8]);
  }
  hs[tid] = (__bf16)0.0f;   // tid < 256 covers all of hs
  float c = 0.0f;
  async_wait0();
  __syncthreads();

  for (int s = 0; s < T; ++s) {
    int t = dir ? (T - 1 - s) : s;
    // matvec: h(256) @ U_half(256x512); wave w owns cols [w*64, w*64+64)
#pragma unroll
    for (int nt = 0; nt < 4; ++nt) {
      v8f acc;
      for (int e = 0; e < 8; ++e) acc[e] = 0.0f;
      int colBase = w * 64 + nt * 16;
#pragma unroll
      for (int kt = 0; kt < 8; ++kt) {
        union { v16bf v; v8bf h[2]; } a, b;
        a.h[0] = *(const v8bf*)&hs[kt * 32 + kg * 8];        // broadcast read
        a.h[1] = *(const v8bf*)&hs[kt * 32 + 16 + kg * 8];
        int col = colBase + (lane & 15);
        b.h[0] = *(const v8bf*)&Us[col * REC_UPITCH + kt * 32 + kg * 8];
        b.h[1] = *(const v8bf*)&Us[col * REC_UPITCH + kt * 32 + 16 + kg * 8];
        acc = __builtin_amdgcn_wmma_f32_16x16x32_bf16(
            false, a.v, false, b.v, (short)0, acc, false, false);
      }
      if (lane < 16) {  // row M=0 of D: acc[0], lanes 0-15
        int col = colBase + lane;
        int g = col >> 7, jj = col & 127;
        zr[col] = acc[0] + zx[(size_t)t * G4 + g * HH + half * 128 + jj];
      }
    }
    __syncthreads();
    // gate math: thread j (<128) owns cell state c_j for the whole sequence
    if (tid < 128) {
      float gi = zr[tid], gf = zr[128 + tid], gg = zr[256 + tid], go = zr[384 + tid];
      c = sigf(gf) * c + sigf(gi) * sigf(gg);
      float h = sigf(go) * sigf(c);
      hs[half * 128 + tid] = (__bf16)h;
      hx[me * 128 + tid]   = (__bf16)h;
      outPlane[(size_t)t * outStride + colOff0 + dir * HH + half * 128 + tid] = h;
      if (outBf)
        outBf[(size_t)t * bfStride + dir * HH + half * 128 + tid] = (__bf16)h;
    }
    // publish own half (release) and acquire partner half
    __threadfence();
    __syncthreads();
    if (tid == 0) {
      atomicAdd(&flags[me], 1u);
      while (atomicAdd(&flags[other], 0u) <= (unsigned)s) __builtin_amdgcn_s_sleep(1);
    }
    __syncthreads();
    __threadfence();
    if (tid < 128) hs[(1 - half) * 128 + tid] = hx[other * 128 + tid];
    __syncthreads();
  }
}

// ---------------------------------------------------------------------------
__global__ void init_flags(unsigned* flags) {
  if (threadIdx.x < 4) flags[threadIdx.x] = 0u;
}

__global__ void copy_x_to_out(const float* __restrict__ x, float* __restrict__ out,
                              size_t planeSize) {
  size_t i = (size_t)blockIdx.x * blockDim.x + threadIdx.x;
  size_t stride = (size_t)gridDim.x * blockDim.x;
  const size_t n = (size_t)TT * DD;
  for (; i < n; i += stride) {
    size_t t = i >> 11, cc = i & 2047;
    float v = x[i];
    out[t * HOUT + cc] = v;
    out[planeSize + t * HOUT + cc] = v;
  }
}

// ---------------------------------------------------------------------------
// Workspace layout (bytes)
// ---------------------------------------------------------------------------
constexpr size_t OFF_XB   = 0;                          // x bf16: 32MB
constexpr size_t OFF_W1F  = OFF_XB   + (size_t)TT*DD*2;
constexpr size_t OFF_W1B  = OFF_W1F  + (size_t)DD*G4*2;
constexpr size_t OFF_W2F  = OFF_W1B  + (size_t)DD*G4*2;
constexpr size_t OFF_W2B  = OFF_W2F  + (size_t)512*G4*2;
constexpr size_t OFF_UP   = OFF_W2B  + (size_t)512*G4*2; // 4 * (2*512*256) bf16
constexpr size_t OFF_ZXA  = OFF_UP   + (size_t)4*2*512*256*2;
constexpr size_t OFF_ZXB  = OFF_ZXA  + (size_t)TT*G4*4;
constexpr size_t OFF_M1BF = OFF_ZXB  + (size_t)TT*G4*4;
constexpr size_t OFF_HX   = OFF_M1BF + (size_t)TT*512*2;
constexpr size_t OFF_FLAG = OFF_HX   + 1024;

extern "C" void kernel_launch(void* const* d_in, const int* in_sizes, int n_in,
                              void* d_out, int out_size, void* d_ws, size_t ws_size,
                              hipStream_t stream) {
  (void)in_sizes; (void)n_in; (void)out_size; (void)ws_size;
  const float* x   = (const float*)d_in[0];
  const float* W1f = (const float*)d_in[1];
  const float* U1f = (const float*)d_in[2];
  const float* b1f = (const float*)d_in[3];
  const float* W1b = (const float*)d_in[4];
  const float* U1b = (const float*)d_in[5];
  const float* b1b = (const float*)d_in[6];
  const float* W2f = (const float*)d_in[7];
  const float* U2f = (const float*)d_in[8];
  const float* b2f = (const float*)d_in[9];
  const float* W2b = (const float*)d_in[10];
  const float* U2b = (const float*)d_in[11];
  const float* b2b = (const float*)d_in[12];
  float* out = (float*)d_out;
  char* ws = (char*)d_ws;

  __bf16* xb   = (__bf16*)(ws + OFF_XB);
  __bf16* w1fb = (__bf16*)(ws + OFF_W1F);
  __bf16* w1bb = (__bf16*)(ws + OFF_W1B);
  __bf16* w2fb = (__bf16*)(ws + OFF_W2F);
  __bf16* w2bb = (__bf16*)(ws + OFF_W2B);
  __bf16* up   = (__bf16*)(ws + OFF_UP);
  float*  zxA  = (float*)(ws + OFF_ZXA);
  float*  zxB  = (float*)(ws + OFF_ZXB);
  __bf16* m1bf = (__bf16*)(ws + OFF_M1BF);
  __bf16* hx   = (__bf16*)(ws + OFF_HX);
  unsigned* fl = (unsigned*)(ws + OFF_FLAG);

  // 1) convert inputs to bf16; W transposed to (N,K); U column-permuted per half
  cvt_f32_to_bf16<<<4096, 256, 0, stream>>>(x, xb, TT * DD);
  cvt_W_transpose<<<2048, 256, 0, stream>>>(W1f, w1fb, DD, G4);
  cvt_W_transpose<<<2048, 256, 0, stream>>>(W1b, w1bb, DD, G4);
  cvt_W_transpose<<< 512, 256, 0, stream>>>(W2f, w2fb, 512, G4);
  cvt_W_transpose<<< 512, 256, 0, stream>>>(W2b, w2bb, 512, G4);
  cvt_U_perm<<<256, 256, 0, stream>>>(U1f, up + (size_t)0 * 2 * 512 * 256);
  cvt_U_perm<<<256, 256, 0, stream>>>(U1b, up + (size_t)1 * 2 * 512 * 256);
  cvt_U_perm<<<256, 256, 0, stream>>>(U2f, up + (size_t)2 * 2 * 512 * 256);
  cvt_U_perm<<<256, 256, 0, stream>>>(U2b, up + (size_t)3 * 2 * 512 * 256);

  dim3 gemmGrid(G4 / GBN, TT / GBM);   // (8, 64)
  const size_t planeSize = (size_t)TT * HOUT;
  const size_t recLds = 512 * REC_UPITCH * 2 + 512 + 512 * 4;  // ~273KB

  // 2) layer-1 input GEMMs:  zx = x @ W + b
  gemm_bf16_wmma<<<gemmGrid, 256, 0, stream>>>(xb, DD, w1fb, DD, b1f, zxA, G4, DD);
  gemm_bf16_wmma<<<gemmGrid, 256, 0, stream>>>(xb, DD, w1bb, DD, b1b, zxB, G4, DD);

  // 3) layer-1 recurrence -> out plane 0 (cols 2048..2559) + M1 bf16
  init_flags<<<1, 64, 0, stream>>>(fl);
  lstm_rec<<<4, 256, recLds, stream>>>(zxA, zxB, up,
                                       out, HOUT, DD,
                                       m1bf, 512, hx, fl, TT);

  // 4) layer-2 input GEMMs (reuse zx buffers):  zx = M1 @ W2 + b
  gemm_bf16_wmma<<<gemmGrid, 256, 0, stream>>>(m1bf, 512, w2fb, 512, b2f, zxA, G4, 512);
  gemm_bf16_wmma<<<gemmGrid, 256, 0, stream>>>(m1bf, 512, w2bb, 512, b2b, zxB, G4, 512);

  // 5) layer-2 recurrence -> out plane 1 (cols 2048..2559)
  init_flags<<<1, 64, 0, stream>>>(fl);
  lstm_rec<<<4, 256, recLds, stream>>>(zxA, zxB, up + (size_t)2 * 2 * 512 * 256,
                                       out + planeSize, HOUT, DD,
                                       (__bf16*)nullptr, 0, hx, fl, TT);

  // 6) x into cols 0..2047 of both planes
  copy_x_to_out<<<4096, 256, 0, stream>>>(x, out, planeSize);
}
